// LocalEncoderWithPooling_34428457844758
// MI455X (gfx1250) — matrix operand order repaired and checked by reading, not compiled, not run on hardware
//
#include <hip/hip_runtime.h>
#include <hip/hip_bf16.h>
#include <stdint.h>
#include <stddef.h>

// Problem dims (fixed by the reference)
#define NUM_B   8
#define SEQ     4096
#define EMB     1024
#define GDIM    2048
#define NPATCH  512
#define MROWS   (NUM_B * NPATCH)   // 4096 GEMM rows

typedef __attribute__((ext_vector_type(16))) __bf16        v16bf;
typedef __attribute__((ext_vector_type(8)))  float         v8f;
typedef __attribute__((ext_vector_type(4)))  float         f32x4;
typedef __attribute__((ext_vector_type(4)))  unsigned int  u32x4;
typedef __attribute__((ext_vector_type(4)))  unsigned short u16x4;

static __device__ __forceinline__ unsigned short f2bf(float f) {
  union { float f; unsigned int u; } c; c.f = f;
  unsigned int u = c.u;
  u += 0x7FFFu + ((u >> 16) & 1u);   // round-to-nearest-even
  return (unsigned short)(u >> 16);
}
static __device__ __forceinline__ float bf2f(unsigned short h) {
  union { float f; unsigned int u; } c; c.u = ((unsigned int)h) << 16;
  return c.f;
}

// ---------------------------------------------------------------------------
// Kernel 1: byte_embeds = bf16(embed_table[bytes])   (bandwidth: 64MB writes)
// One block per token, 256 threads x float4 = 1024 elements.
// ---------------------------------------------------------------------------
__global__ void k_embed(const int* __restrict__ bytes,
                        const float* __restrict__ table,
                        unsigned short* __restrict__ out0) {
  int tok = blockIdx.x;                 // [0, B*S)
  int byte = bytes[tok];                // wave-uniform -> s_load
  const f32x4* src = (const f32x4*)(table + (size_t)byte * EMB);
  u16x4* dst = (u16x4*)(out0 + (size_t)tok * EMB);
  f32x4 v = src[threadIdx.x];
  u16x4 o;
  o.x = f2bf(v.x); o.y = f2bf(v.y); o.z = f2bf(v.z); o.w = f2bf(v.w);
  dst[threadIdx.x] = o;
}

// ---------------------------------------------------------------------------
// Kernel 2: patch mean pooling via binary search on sorted patch_ids.
// One block per (batch, patch); no atomics. Writes bf16 A-matrix rows for the
// GEMM and the patch mask (row max == last element since sorted).
// ---------------------------------------------------------------------------
__global__ void k_pool(const int* __restrict__ bytes,
                       const int* __restrict__ pids,
                       const float* __restrict__ table,
                       unsigned short* __restrict__ Abf,   // [MROWS][EMB] bf16
                       float* __restrict__ mask) {
  int bp = blockIdx.x;                  // [0, B*NPATCH)
  int b  = bp >> 9;                     // / 512
  int p  = bp & (NPATCH - 1);
  const int* ids = pids + (size_t)b * SEQ;

  // lower_bound(p) and upper_bound(p) — uniform scalar search
  int lo = 0, hi = SEQ;
  while (lo < hi) { int m = (lo + hi) >> 1; if (ids[m] <  p) lo = m + 1; else hi = m; }
  int start = lo;
  hi = SEQ;
  while (lo < hi) { int m = (lo + hi) >> 1; if (ids[m] <= p) lo = m + 1; else hi = m; }
  int end = lo;
  int cnt = end - start;

  int col = threadIdx.x * 4;            // 256 threads x 4 = 1024 cols
  const int* bb = bytes + (size_t)b * SEQ;
  f32x4 acc = (f32x4)0.0f;
  for (int t = start; t < end; ++t) {
    int byte = bb[t];                   // uniform -> scalar load
    f32x4 v = *(const f32x4*)(table + (size_t)byte * EMB + col);
    // reference gathers in bf16; round through bf16 before accumulating
    acc.x += bf2f(f2bf(v.x)); acc.y += bf2f(f2bf(v.y));
    acc.z += bf2f(f2bf(v.z)); acc.w += bf2f(f2bf(v.w));
  }
  float inv = (cnt > 0) ? 1.0f / (float)cnt : 0.0f;
  u16x4 o;
  o.x = f2bf(acc.x * inv); o.y = f2bf(acc.y * inv);
  o.z = f2bf(acc.z * inv); o.w = f2bf(acc.w * inv);
  *(u16x4*)(Abf + (size_t)bp * EMB + col) = o;

  if (threadIdx.x == 0) mask[bp] = (p <= ids[SEQ - 1]) ? 1.0f : 0.0f;
}

// ---------------------------------------------------------------------------
// Kernel 3: pack proj_w (ExG f32 row-major) into per-lane WMMA-B bf16
// fragments so the GEMM streams them with coalesced b128 loads.
// B fragment (v_wmma_f32_16x16x32_bf16): lane L holds column N = n0+(L&15),
// K = base..base+7 and base+16..base+23, base = k0 + ((L<16)?0:8) (16 bf16 = 32B per lane).
// Bpack index: (((ntile*(EMB/32)) + kblk)*32 + lane) * 16 ushorts.
// ---------------------------------------------------------------------------
__global__ void k_packB(const float* __restrict__ W,
                        unsigned short* __restrict__ Bp) {
  __shared__ unsigned short tile[32 * 16];     // K x N tile as bf16
  int ntile = blockIdx.x;                      // [0, GDIM/16)
  int kblk  = blockIdx.y;                      // [0, EMB/32)
  int n0 = ntile * 16, k0 = kblk * 32;
  for (int e = threadIdx.x; e < 512; e += blockDim.x) {
    int r = e >> 4, c = e & 15;
    tile[e] = f2bf(W[(size_t)(k0 + r) * GDIM + n0 + c]);
  }
  __syncthreads();
  if (threadIdx.x < 32) {
    int L = threadIdx.x;
    int c  = L & 15;
    int kb = (L < 16) ? 0 : 8;
    unsigned short frag[16];
    #pragma unroll
    for (int j = 0; j < 16; ++j) frag[j] = tile[(kb + j + (j >= 8 ? 8 : 0)) * 16 + c];
    u32x4* dst = (u32x4*)(Bp + ((((size_t)ntile * (EMB / 32) + kblk) * 32 + L) * 16));
    const u32x4* s = (const u32x4*)frag;
    dst[0] = s[0]; dst[1] = s[1];
  }
}

// ---------------------------------------------------------------------------
// Kernel 4: C[4096x2048] = A(bf16) x W(bf16) + bias, fp32 accumulate via
// v_wmma_f32_16x16x32_bf16. Block = 8 waves; all waves share one 16-row A
// tile staged in padded LDS (row stride 1032 ushorts -> bank-spread b128
// reads). Each wave owns one 16x16 C tile, K-loop of 32.
// A fragment per lane: M = L&15, K = base..base+7 and base+16..base+23,
// base = k0 + ((L<16)?0:8)   (ISA 7.12.2, 16-bit A 16x32 layout).
// ---------------------------------------------------------------------------
#define ASTRIDE (EMB + 8)
__global__ void k_gemm(const unsigned short* __restrict__ A,   // [MROWS][EMB]
                       const unsigned short* __restrict__ Bp,  // packed frags
                       const float* __restrict__ bias,
                       float* __restrict__ C) {
  __shared__ unsigned short Atile[16 * ASTRIDE];
  int mBase = blockIdx.y * 16;

  // stage 16 x 1024 bf16 (32KB) cooperatively, 256 thr x 8 x b128
  for (int i = threadIdx.x; i < 16 * (EMB / 8); i += blockDim.x) {
    int r  = i >> 7;          // / 128
    int c8 = i & 127;         // uint4 within row
    u32x4 v = *(const u32x4*)(A + (size_t)(mBase + r) * EMB + c8 * 8);
    *(u32x4*)(&Atile[r * ASTRIDE + c8 * 8]) = v;
  }
  __syncthreads();

  int wave = threadIdx.x >> 5;
  int lane = threadIdx.x & 31;
  int ntile = blockIdx.x * 8 + wave;
  int nBase = ntile * 16;

  int rowA   = lane & 15;
  int kHalfA = (lane < 16) ? 0 : 8;
  const unsigned short* arow = &Atile[rowA * ASTRIDE];
  const unsigned short* bbase = Bp + ((size_t)ntile * (EMB / 32) * 32 + lane) * 16;

  union Frag { v16bf v; u32x4 q[2]; };
  v8f acc = (v8f)0.0f;

  for (int k0 = 0; k0 < EMB; k0 += 32) {
    Frag a, bfr;
    a.q[0] = *(const u32x4*)(arow + k0 + kHalfA);        // ds_load_b128
    a.q[1] = *(const u32x4*)(arow + k0 + kHalfA + 16);
    const u32x4* bp = (const u32x4*)(bbase + (size_t)(k0 >> 5) * 32 * 16);
    bfr.q[0] = bp[0];                                     // coalesced 32B/lane
    bfr.q[1] = bp[1];
    acc = __builtin_amdgcn_wmma_f32_16x16x32_bf16(
        false, a.v, false, bfr.v, (short)0, acc, false, false);
  }

  // C/D layout: VGPR r -> (M = mBase + r [+8 for lanes 16..31], N = nBase + (lane&15))
  float bv = bias[nBase + (lane & 15)];
  int colC = nBase + (lane & 15);
  int rOff = (lane < 16) ? 0 : 8;
  #pragma unroll
  for (int r = 0; r < 8; ++r)
    C[(size_t)(mBase + rOff + r) * GDIM + colC] = acc[r] + bv;
}

// ---------------------------------------------------------------------------
extern "C" void kernel_launch(void* const* d_in, const int* in_sizes, int n_in,
                              void* d_out, int out_size, void* d_ws, size_t ws_size,
                              hipStream_t stream) {
  const int*   bytes = (const int*)d_in[0];
  const int*   pids  = (const int*)d_in[1];
  const float* table = (const float*)d_in[2];
  const float* W     = (const float*)d_in[3];
  const float* bias  = (const float*)d_in[4];

  // Output tuple flattened in return order:
  //   [0] byte_embeds bf16 (B*S*E), [1] patch_embs f32 (B*P*G), [2] mask f32 (B*P)
  char* out = (char*)d_out;
  unsigned short* out_embeds = (unsigned short*)out;
  float* out_patch = (float*)(out + (size_t)NUM_B * SEQ * EMB * 2);
  float* out_mask  = (float*)(out + (size_t)NUM_B * SEQ * EMB * 2
                                  + (size_t)MROWS * GDIM * 4);

  // Workspace: A (8MB bf16) then packed-B (4MB bf16)
  unsigned short* Abf = (unsigned short*)d_ws;
  unsigned short* Bp  = (unsigned short*)((char*)d_ws + (size_t)MROWS * EMB * 2);

  k_embed<<<NUM_B * SEQ, 256, 0, stream>>>(bytes, table, out_embeds);
  k_pool <<<NUM_B * NPATCH, 256, 0, stream>>>(bytes, pids, table, Abf, out_mask);
  dim3 gp(GDIM / 16, EMB / 32);
  k_packB<<<gp, 128, 0, stream>>>(W, Bp);
  dim3 gg(GDIM / 128, MROWS / 16);   // 8 waves/block, one 16x16 tile each
  k_gemm<<<gg, 256, 0, stream>>>(Abf, Bp, bias, out_patch);
}
